// Decoder_84997402788281
// MI455X (gfx1250) — compile-verified, matrix operand order checked by reading, not run on hardware
//
#include <hip/hip_runtime.h>
#include <math.h>

#define B_ 64
#define S_ 8
#define LC_ 64
#define LU_ 512
#define LG_ 64
#define H_ 1024
#define E_ 512
#define V_ 32000
#define OOV_ 50
#define VO_ (V_ + OOV_)

typedef __bf16 bf16_t;
typedef __attribute__((ext_vector_type(16))) __bf16   v16bf;
typedef __attribute__((ext_vector_type(8)))  float    v8f;
typedef __attribute__((ext_vector_type(8)))  unsigned v8u;

static __device__ __forceinline__ unsigned pk2bf(float hi, float lo) {
  // v_perm_b32: pack high halves of two f32 -> {bf16(hi),bf16(lo)} (truncating round)
  return __builtin_amdgcn_perm(__builtin_bit_cast(unsigned, hi),
                               __builtin_bit_cast(unsigned, lo), 0x07060302u);
}

static __device__ __forceinline__ float fast_tanh(float x) {
  float y;
  asm("v_tanh_f32 %0, %1" : "=v"(y) : "v"(x));
  return y;
}

// Load a 16x32 bf16 fragment row (A) or col (B) for one lane.
// ISA layout (16-bit 16x32 A): lanes 0-15 hold K={0..7,16..23}, lanes 16-31 K={8..15,24..31}.
static __device__ __forceinline__ v16bf load_frag(const float* __restrict__ rowbase,
                                                  int k0, int lane) {
  int kb = k0 + ((lane & 16) ? 8 : 0);
  const float4* p = reinterpret_cast<const float4*>(rowbase + kb);
  float4 x0 = p[0], x1 = p[1], x2 = p[4], x3 = p[5];
  v8u u;
  u[0] = pk2bf(x0.y, x0.x);
  u[1] = pk2bf(x0.w, x0.z);
  u[2] = pk2bf(x1.y, x1.x);
  u[3] = pk2bf(x1.w, x1.z);
  u[4] = pk2bf(x2.y, x2.x);
  u[5] = pk2bf(x2.w, x2.z);
  u[6] = pk2bf(x3.y, x3.x);
  u[7] = pk2bf(x3.w, x3.z);
  return __builtin_bit_cast(v16bf, u);
}

// C[64,N] = A[64,K] * W[N,K]^T + bias.  K mult of 32, N mult of 16, M fixed at 64.
// One wave per 16-wide N tile; each wave computes all 4 M tiles so the weight
// fragment (the big operand) is loaded+converted once per 4 WMMAs.
__global__ void __launch_bounds__(256)
gemm_wmma_kernel(const float* __restrict__ A, const float* __restrict__ W,
                 const float* __restrict__ bias, float* __restrict__ C,
                 int K, int N) {
  int lane = threadIdx.x & 31;
  int wave = threadIdx.x >> 5;
  int nt = blockIdx.x * (blockDim.x >> 5) + wave;
  if (nt >= (N >> 4)) return;
  const float* wrow = W + (size_t)(nt * 16 + (lane & 15)) * K;
  const float* arow = A + (size_t)(lane & 15) * K;
  v8f acc0 = {}, acc1 = {}, acc2 = {}, acc3 = {};
  for (int k0 = 0; k0 < K; k0 += 32) {
    __builtin_prefetch(wrow + k0 + 64, 0, 1);
    v16bf b = load_frag(wrow, k0, lane);
    v16bf a0 = load_frag(arow, k0, lane);
    v16bf a1 = load_frag(arow + (size_t)16 * K, k0, lane);
    v16bf a2 = load_frag(arow + (size_t)32 * K, k0, lane);
    v16bf a3 = load_frag(arow + (size_t)48 * K, k0, lane);
    acc0 = __builtin_amdgcn_wmma_f32_16x16x32_bf16(false, a0, false, b, (short)0, acc0, false, false);
    acc1 = __builtin_amdgcn_wmma_f32_16x16x32_bf16(false, a1, false, b, (short)0, acc1, false, false);
    acc2 = __builtin_amdgcn_wmma_f32_16x16x32_bf16(false, a2, false, b, (short)0, acc2, false, false);
    acc3 = __builtin_amdgcn_wmma_f32_16x16x32_bf16(false, a3, false, b, (short)0, acc3, false, false);
  }
  int col = nt * 16 + (lane & 15);
  float bv = bias ? bias[col] : 0.0f;
  int rbase = ((lane & 16) ? 8 : 0);
#pragma unroll
  for (int r = 0; r < 8; ++r) {
    C[(size_t)(rbase + r) * N + col]      = acc0[r] + bv;
    C[(size_t)(rbase + r + 16) * N + col] = acc1[r] + bv;
    C[(size_t)(rbase + r + 32) * N + col] = acc2[r] + bv;
    C[(size_t)(rbase + r + 48) * N + col] = acc3[r] + bv;
  }
}

// Fused additive-attention scores:
// scores[row] = sum_h v[h] * tanh( (mem[row]·Wm[h]) + aq[b(row), h] ), with mask.
// One wave per 16 rows; h processed in chunks of 8 WMMA tiles (128 h) so the
// projected [R,1024] matrix is never materialized; A fragment reused 8x.
__global__ void __launch_bounds__(256)
attn_score_kernel(const float* __restrict__ mem, const float* __restrict__ Wm,
                  const float* __restrict__ aq, const float* __restrict__ vvec,
                  const unsigned char* __restrict__ mask,
                  float* __restrict__ scores, int R, int rows_per_b) {
  const int K = H_;
  int lane = threadIdx.x & 31;
  int wave = threadIdx.x >> 5;
  int t = blockIdx.x * (blockDim.x >> 5) + wave;
  int row0 = t * 16;
  if (row0 >= R) return;
  const float* arow = mem + (size_t)(row0 + (lane & 15)) * K;
  int half8 = ((lane & 16) ? 8 : 0);
  float sp[8];
#pragma unroll
  for (int r = 0; r < 8; ++r) sp[r] = 0.f;
  v8f vzero = {};
  for (int hc = 0; hc < H_; hc += 128) {
    v8f acc[8];
#pragma unroll
    for (int j = 0; j < 8; ++j) acc[j] = vzero;
    for (int k0 = 0; k0 < K; k0 += 32) {
      __builtin_prefetch(arow + k0 + 64, 0, 1);
      v16bf a = load_frag(arow, k0, lane);
#pragma unroll
      for (int j = 0; j < 8; ++j) {
        const float* wrow = Wm + (size_t)(hc + j * 16 + (lane & 15)) * K;
        v16bf b = load_frag(wrow, k0, lane);
        acc[j] = __builtin_amdgcn_wmma_f32_16x16x32_bf16(false, a, false, b,
                                                         (short)0, acc[j], false, false);
      }
    }
    int hbase = hc + (lane & 15);
#pragma unroll
    for (int j = 0; j < 8; ++j) {
      int hj = hbase + j * 16;
      float vv = vvec[hj];
#pragma unroll
      for (int r = 0; r < 8; ++r) {
        int row = row0 + half8 + r;
        int b = row / rows_per_b;
        float val = fast_tanh(acc[j][r] + aq[(size_t)b * H_ + hj]);
        sp[r] += vv * val;
      }
    }
  }
  // reduce across the 16 lanes of each half-wave (column reduction)
#pragma unroll
  for (int r = 0; r < 8; ++r) {
    sp[r] += __shfl_xor(sp[r], 8);
    sp[r] += __shfl_xor(sp[r], 4);
    sp[r] += __shfl_xor(sp[r], 2);
    sp[r] += __shfl_xor(sp[r], 1);
  }
  if ((lane & 15) == 0) {
#pragma unroll
    for (int r = 0; r < 8; ++r) {
      int row = row0 + half8 + r;
      float sc = sp[r];
      if (mask && mask[row]) sc = -1.0e9f;
      scores[row] = sc;
    }
  }
}

// Per-group softmax over L scores (in-place -> weights) then ctx[g] = sum_l w_l * mem[g,l,:]
__global__ void __launch_bounds__(256)
softmax_ctx_kernel(float* __restrict__ wbuf, const float* __restrict__ mem,
                   float* __restrict__ ctx, int L, int ctx_stride) {
  __shared__ float red[256];
  __shared__ float wl[512];
  int g = blockIdx.x, tid = threadIdx.x;
  float* sc = wbuf + (size_t)g * L;
  float lm = -3.0e38f;
  for (int l = tid; l < L; l += 256) lm = fmaxf(lm, sc[l]);
  red[tid] = lm; __syncthreads();
  for (int s = 128; s > 0; s >>= 1) { if (tid < s) red[tid] = fmaxf(red[tid], red[tid + s]); __syncthreads(); }
  float mx = red[0]; __syncthreads();
  float ls = 0.f;
  for (int l = tid; l < L; l += 256) { float e = __expf(sc[l] - mx); wl[l] = e; ls += e; }
  red[tid] = ls; __syncthreads();
  for (int s = 128; s > 0; s >>= 1) { if (tid < s) red[tid] += red[tid + s]; __syncthreads(); }
  float inv = 1.0f / red[0]; __syncthreads();
  for (int l = tid; l < L; l += 256) { wl[l] *= inv; sc[l] = wl[l]; }
  __syncthreads();
  const float* mbase = mem + (size_t)g * L * H_;
  for (int h = tid; h < H_; h += 256) {
    float acc = 0.f;
    for (int l = 0; l < L; ++l) acc += wl[l] * mbase[(size_t)l * H_ + h];
    ctx[(size_t)g * ctx_stride + h] = acc;
  }
}

__global__ void cue_combine_kernel(const float* __restrict__ kg, const float* __restrict__ cue_bs,
                                   float* __restrict__ ctx_stack) {
  int b = blockIdx.x;
  for (int h = threadIdx.x; h < H_; h += blockDim.x) {
    float acc = 0.f;
    for (int s = 0; s < S_; ++s)
      acc += kg[b * S_ + s] * cue_bs[(size_t)(b * S_ + s) * H_ + h];
    ctx_stack[(size_t)b * 3 * H_ + H_ + h] = acc;
  }
}

__global__ void concat_x_kernel(const float* __restrict__ emb, const float* __restrict__ dic,
                                float* __restrict__ x) {
  int b = blockIdx.x;
  for (int k = threadIdx.x; k < E_ + H_; k += blockDim.x)
    x[(size_t)b * (E_ + H_) + k] = (k < E_) ? emb[b * E_ + k] : dic[(size_t)b * H_ + (k - E_)];
}

__global__ void gru_gates_kernel(const float* __restrict__ gi, const float* __restrict__ gh,
                                 const float* __restrict__ h0, float* __restrict__ h) {
  int idx = blockIdx.x * blockDim.x + threadIdx.x;
  if (idx >= B_ * H_) return;
  int b = idx / H_, j = idx % H_;
  const float* gib = gi + (size_t)b * 3 * H_;
  const float* ghb = gh + (size_t)b * 3 * H_;
  float r = 1.f / (1.f + __expf(-(gib[j] + ghb[j])));
  float z = 1.f / (1.f + __expf(-(gib[H_ + j] + ghb[H_ + j])));
  float n = fast_tanh(gib[2 * H_ + j] + r * ghb[2 * H_ + j]);
  h[idx] = (1.f - z) * n + z * h0[idx];
}

__global__ void concat_outin_kernel(const float* __restrict__ h, const float* __restrict__ cm,
                                    float* __restrict__ o) {
  int b = blockIdx.x;
  for (int k = threadIdx.x; k < 2 * H_; k += blockDim.x)
    o[(size_t)b * 2 * H_ + k] = (k < H_) ? h[(size_t)b * H_ + k] : cm[(size_t)b * H_ + (k - H_)];
}

__global__ void pgen_kernel(const float* __restrict__ emb, const float* __restrict__ h,
                            const float* __restrict__ cm, const float* __restrict__ W,
                            const float* __restrict__ bias, float* __restrict__ pg) {
  __shared__ float red[256];
  int b = blockIdx.x, tid = threadIdx.x;
  float acc = 0.f;
  for (int k = tid; k < E_ + 2 * H_; k += 256) {
    float x;
    if (k < E_)           x = emb[b * E_ + k];
    else if (k < E_ + H_) x = h[(size_t)b * H_ + (k - E_)];
    else                  x = cm[(size_t)b * H_ + (k - E_ - H_)];
    acc += x * W[k];
  }
  red[tid] = acc; __syncthreads();
  for (int s = 128; s > 0; s >>= 1) { if (tid < s) red[tid] += red[tid + s]; __syncthreads(); }
  if (tid == 0) pg[b] = 1.f / (1.f + __expf(-(red[0] + bias[0])));
}

__global__ void __launch_bounds__(256)
vocab_softmax_kernel(const float* __restrict__ logits, const float* __restrict__ pg,
                     float* __restrict__ out) {
  __shared__ float red[256];
  int b = blockIdx.x, tid = threadIdx.x;
  const float* lr = logits + (size_t)b * V_;
  float lm = -3.0e38f;
  for (int i = tid; i < V_; i += 256) lm = fmaxf(lm, lr[i]);
  red[tid] = lm; __syncthreads();
  for (int s = 128; s > 0; s >>= 1) { if (tid < s) red[tid] = fmaxf(red[tid], red[tid + s]); __syncthreads(); }
  float mx = red[0]; __syncthreads();
  float ls = 0.f;
  for (int i = tid; i < V_; i += 256) ls += __expf(lr[i] - mx);
  red[tid] = ls; __syncthreads();
  for (int s = 128; s > 0; s >>= 1) { if (tid < s) red[tid] += red[tid + s]; __syncthreads(); }
  float scale = pg[b] / red[0];
  float* orow = out + (size_t)b * VO_;
  for (int i = tid; i < V_; i += 256) orow[i] = __expf(lr[i] - mx) * scale;
  for (int i = V_ + tid; i < VO_; i += 256) orow[i] = 0.f;
}

__global__ void scatter_kernel(const float* __restrict__ wu, const float* __restrict__ wc,
                               const float* __restrict__ wg, const float* __restrict__ ctw,
                               const float* __restrict__ kg, const float* __restrict__ pg,
                               const int* __restrict__ iu, const int* __restrict__ ic,
                               const int* __restrict__ ig, float* __restrict__ out) {
  const int PER = LU_ + S_ * LC_ + LG_;
  int i = blockIdx.x * blockDim.x + threadIdx.x;
  if (i >= B_ * PER) return;
  int b = i / PER, j = i % PER;
  float g1 = 1.0f - pg[b];
  float val; int idx;
  if (j < LU_) {
    val = ctw[b * 3 + 0] * wu[b * LU_ + j];
    idx = iu[b * LU_ + j];
  } else if (j < LU_ + S_ * LC_) {
    int m = j - LU_; int s = m / LC_;
    val = ctw[b * 3 + 1] * kg[b * S_ + s] * wc[(size_t)b * S_ * LC_ + m];
    idx = ic[(size_t)b * S_ * LC_ + m];
  } else {
    int l = j - LU_ - S_ * LC_;
    val = ctw[b * 3 + 2] * wg[b * LG_ + l];
    idx = ig[b * LG_ + l];
  }
  atomicAdd(&out[(size_t)b * VO_ + idx], g1 * val);
}

extern "C" void kernel_launch(void* const* d_in, const int* in_sizes, int n_in,
                              void* d_out, int out_size, void* d_ws, size_t ws_size,
                              hipStream_t stream) {
  (void)in_sizes; (void)n_in; (void)out_size; (void)ws_size;
  const float* input_emb  = (const float*)d_in[0];
  const float* dec_hidden = (const float*)d_in[1];   // [1,B,H] -> h0
  const float* dec_init_c = (const float*)d_in[2];
  const float* utt_out    = (const float*)d_in[3];
  const float* cue_out    = (const float*)d_in[4];
  const float* goal_out   = (const float*)d_in[5];
  const float* kg         = (const float*)d_in[6];
  const unsigned char* utt_mask  = (const unsigned char*)d_in[7];
  const unsigned char* cue_mask  = (const unsigned char*)d_in[8];
  const unsigned char* goal_mask = (const unsigned char*)d_in[9];
  const int* iu = (const int*)d_in[10];
  const int* ic = (const int*)d_in[11];
  const int* ig = (const int*)d_in[12];
  const float* w_ih = (const float*)d_in[14];
  const float* w_hh = (const float*)d_in[15];
  const float* b_ih = (const float*)d_in[16];
  const float* b_hh = (const float*)d_in[17];
  const float* pgen_W = (const float*)d_in[18];
  const float* pgen_b = (const float*)d_in[19];
  const float* out1_W = (const float*)d_in[20];
  const float* out1_b = (const float*)d_in[21];
  const float* out2_W = (const float*)d_in[22];
  const float* out2_b = (const float*)d_in[23];
  const float* utt_Wq = (const float*)d_in[24]; const float* utt_bq = (const float*)d_in[25];
  const float* utt_Wm = (const float*)d_in[26]; const float* utt_v  = (const float*)d_in[27];
  const float* cue_Wq = (const float*)d_in[28]; const float* cue_bq = (const float*)d_in[29];
  const float* cue_Wm = (const float*)d_in[30]; const float* cue_v  = (const float*)d_in[31];
  const float* goal_Wq = (const float*)d_in[32]; const float* goal_bq = (const float*)d_in[33];
  const float* goal_Wm = (const float*)d_in[34]; const float* goal_v  = (const float*)d_in[35];
  const float* hl_Wq = (const float*)d_in[36]; const float* hl_bq = (const float*)d_in[37];
  const float* hl_Wm = (const float*)d_in[38]; const float* hl_v  = (const float*)d_in[39];
  float* out = (float*)d_out;
  float* ws  = (float*)d_ws;

  // workspace layout (floats)
  float* x_cat     = ws;                           // 64*1536
  float* gi        = x_cat + 64 * 1536;            // 64*3072
  float* gh        = gi + 64 * 3072;               // 64*3072
  float* h         = gh + 64 * 3072;               // 64*1024
  float* a_utt     = h + 64 * 1024;                // 64*1024 x4
  float* a_cue     = a_utt + 64 * 1024;
  float* a_goal    = a_cue + 64 * 1024;
  float* a_hl      = a_goal + 64 * 1024;
  float* sc_utt    = a_hl + 64 * 1024;             // 32768
  float* sc_cue    = sc_utt + B_ * LU_;            // 32768
  float* sc_goal   = sc_cue + B_ * S_ * LC_;       // 4096
  float* sc_hl     = sc_goal + B_ * LG_;           // 192 (pad 256)
  float* ctx_stack = sc_hl + 256;                  // 64*3*1024
  float* cue_bs    = ctx_stack + (size_t)B_ * 3 * H_;   // 64*8*1024
  float* ctx_merge = cue_bs + (size_t)B_ * S_ * H_;     // 64*1024
  float* p_gen     = ctx_merge + (size_t)B_ * H_;       // 64
  float* out_in    = p_gen + 64;                        // 64*2048
  float* hid1      = out_in + (size_t)B_ * 2 * H_;      // 64*1024
  float* logits    = hid1 + (size_t)B_ * H_;            // 64*32000

  // 1) GRU
  concat_x_kernel<<<B_, 256, 0, stream>>>(input_emb, dec_init_c, x_cat);
  gemm_wmma_kernel<<<24, 256, 0, stream>>>(x_cat, w_ih, b_ih, gi, E_ + H_, 3 * H_);
  gemm_wmma_kernel<<<24, 256, 0, stream>>>(dec_hidden, w_hh, b_hh, gh, H_, 3 * H_);
  gru_gates_kernel<<<(B_ * H_) / 256, 256, 0, stream>>>(gi, gh, dec_hidden, h);

  // 2) query projections a = h @ Wq^T + bq  (64x1024 each)
  gemm_wmma_kernel<<<8, 256, 0, stream>>>(h, utt_Wq, utt_bq, a_utt, H_, H_);
  gemm_wmma_kernel<<<8, 256, 0, stream>>>(h, cue_Wq, cue_bq, a_cue, H_, H_);
  gemm_wmma_kernel<<<8, 256, 0, stream>>>(h, goal_Wq, goal_bq, a_goal, H_, H_);
  gemm_wmma_kernel<<<8, 256, 0, stream>>>(h, hl_Wq, hl_bq, a_hl, H_, H_);

  // 3) fused attention scores (WMMA + tanh + v-dot)
  attn_score_kernel<<<256, 256, 0, stream>>>(utt_out, utt_Wm, a_utt, utt_v, utt_mask,
                                             sc_utt, B_ * LU_, LU_);
  attn_score_kernel<<<256, 256, 0, stream>>>(cue_out, cue_Wm, a_cue, cue_v, cue_mask,
                                             sc_cue, B_ * S_ * LC_, S_ * LC_);
  attn_score_kernel<<<32, 256, 0, stream>>>(goal_out, goal_Wm, a_goal, goal_v, goal_mask,
                                            sc_goal, B_ * LG_, LG_);

  // 4) softmax + ctx (utt/goal write straight into ctx_stack slots 0/2)
  softmax_ctx_kernel<<<B_, 256, 0, stream>>>(sc_utt, utt_out, ctx_stack + 0, LU_, 3 * H_);
  softmax_ctx_kernel<<<B_ * S_, 256, 0, stream>>>(sc_cue, cue_out, cue_bs, LC_, H_);
  softmax_ctx_kernel<<<B_, 256, 0, stream>>>(sc_goal, goal_out, ctx_stack + 2 * H_, LG_, 3 * H_);
  cue_combine_kernel<<<B_, 256, 0, stream>>>(kg, cue_bs, ctx_stack);

  // 5) hierarchical attention over ctx_stack [B,3,H]
  attn_score_kernel<<<2, 256, 0, stream>>>(ctx_stack, hl_Wm, a_hl, hl_v, nullptr,
                                           sc_hl, B_ * 3, 3);
  softmax_ctx_kernel<<<B_, 256, 0, stream>>>(sc_hl, ctx_stack, ctx_merge, 3, H_);

  // 6) p_gen, output projections
  pgen_kernel<<<B_, 256, 0, stream>>>(input_emb, h, ctx_merge, pgen_W, pgen_b, p_gen);
  concat_outin_kernel<<<B_, 256, 0, stream>>>(h, ctx_merge, out_in);
  gemm_wmma_kernel<<<8, 256, 0, stream>>>(out_in, out1_W, out1_b, hid1, 2 * H_, H_);
  gemm_wmma_kernel<<<250, 256, 0, stream>>>(hid1, out2_W, out2_b, logits, H_, V_);

  // 7) vocab softmax * p_gen (+ OOV zeros) then scatter-add copy distribution
  vocab_softmax_kernel<<<B_, 256, 0, stream>>>(logits, p_gen, out);
  int total = B_ * (LU_ + S_ * LC_ + LG_);
  scatter_kernel<<<(total + 255) / 256, 256, 0, stream>>>(sc_utt, sc_cue, sc_goal, sc_hl,
                                                          kg, p_gen, iu, ic, ig, out);
}
